// BiLSTMGenerator_88639535055033
// MI455X (gfx1250) — matrix operand order repaired and checked by reading, not compiled, not run on hardware
//
#include <hip/hip_runtime.h>

typedef __attribute__((ext_vector_type(16))) _Float16 v16h;
typedef __attribute__((ext_vector_type(8)))  _Float16 v8h;
typedef __attribute__((ext_vector_type(8)))  float    v8f;

#define TT    216
#define BB    16384
#define INDIM 8
#define HS1   16
#define HS2   64

__device__ __forceinline__ float fast_sigmoid(float x) {
  // 1/(1+exp(-x)) via exp2 + rcp (v_exp_f32 / v_rcp_f32)
  float e = __builtin_amdgcn_exp2f(-1.44269504088896340736f * x);
  return __builtin_amdgcn_rcpf(1.0f + e);
}
__device__ __forceinline__ float fast_tanh(float x) {
  // tanh(x) = 2*sigmoid(2x) - 1
  float e = __builtin_amdgcn_exp2f(-2.88539008177792681472f * x);
  return 2.0f * __builtin_amdgcn_rcpf(1.0f + e) - 1.0f;
}

// ---------------------------------------------------------------------------
// Small bidirectional-leg LSTM: IN=8 -> H=16, gates = 64 = 4 WMMA N-tiles.
// One wave32 owns a 16-batch tile for all T steps. K = [x(8)|h(16)|pad(8)] = 32.
// Writes hidden states as f16 to hs[T,B,16] for the middle layer.
// ---------------------------------------------------------------------------
__global__ __launch_bounds__(256) void lstm16_dir(
    const float* __restrict__ x,
    const float* __restrict__ h0, const float* __restrict__ c0,
    const float* __restrict__ Wih, const float* __restrict__ Whh,
    const float* __restrict__ bih, const float* __restrict__ bhh,
    _Float16* __restrict__ hs, int reverse)
{
  __shared__ __align__(32) _Float16 sh[8 * 16 * HS1];  // per-wave h staging
  const int tid  = threadIdx.x;
  const int wave = tid >> 5;
  const int lane = tid & 31;
  const int lo   = lane & 15;   // N / batch-row selector
  const int hi   = lane >> 4;   // half-wave selector
  const int baseb = blockIdx.x * 128 + wave * 16;
  _Float16* shh = sh + wave * 16 * HS1;

  // --- B operand (weights), ISA B-layout: lane L element e holds K = (e<8 ? e : e+8) + 8*(L>>4),
  //     N = nt*16 + (L&15). K: 0..7 = Wih cols, 8..23 = Whh cols, 24..31 = zero pad.
  v16h Bw[4];
#pragma unroll
  for (int nt = 0; nt < 4; ++nt) {
    const int N = nt * 16 + lo;
#pragma unroll
    for (int e = 0; e < 16; ++e) {
      const int K = (e < 8 ? e : e + 8) + 8 * hi;
      float w = 0.0f;
      if (K < 8)       w = Wih[N * INDIM + K];
      else if (K < 24) w = Whh[N * HS1 + (K - 8)];
      Bw[nt][e] = (_Float16)w;
    }
  }
  float bias[4];
#pragma unroll
  for (int nt = 0; nt < 4; ++nt) bias[nt] = bih[nt * 16 + lo] + bhh[nt * 16 + lo];

  // --- c state in C/D accumulator layout: VGPR r, lane L -> batch r+8*(L>>4), j = L&15
  v8f c;
#pragma unroll
  for (int r = 0; r < 8; ++r)
    c[r] = c0[(size_t)(baseb + r + 8 * hi) * HS1 + lo];
  // --- h0 -> LDS staging (f16), [b_local*16 + j]
#pragma unroll
  for (int r = 0; r < 8; ++r) {
    const int bl = r + 8 * hi;
    shh[bl * HS1 + lo] = (_Float16)h0[(size_t)(baseb + bl) * HS1 + lo];
  }
  __syncthreads();

  for (int s = 0; s < TT; ++s) {
    const int t = reverse ? (TT - 1 - s) : s;
    // --- A operand, ISA A-layout for 16x32 f16:
    //  lanes 0-15 : e0..7 -> K0..7  (= x[0..7]),  e8..15 -> K16..23 (= h[8..15])
    //  lanes 16-31: e0..7 -> K8..15 (= h[0..7]),  e8..15 -> K24..31 (= zero pad)
    union { v16h v; v8h p[2]; } A;
    if (hi == 0) {
      const float4* xp = (const float4*)(x + ((size_t)t * BB + baseb + lo) * INDIM);
      const float4 x0 = xp[0], x1 = xp[1];
      A.v[0] = (_Float16)x0.x; A.v[1] = (_Float16)x0.y;
      A.v[2] = (_Float16)x0.z; A.v[3] = (_Float16)x0.w;
      A.v[4] = (_Float16)x1.x; A.v[5] = (_Float16)x1.y;
      A.v[6] = (_Float16)x1.z; A.v[7] = (_Float16)x1.w;
      A.p[1] = *(const v8h*)(shh + lo * HS1 + 8);
    } else {
      A.p[0] = *(const v8h*)(shh + lo * HS1);
#pragma unroll
      for (int e = 8; e < 16; ++e) A.v[e] = (_Float16)0.0f;
    }

    // --- gates: 4 independent 16x16x32 WMMAs (tiles = i, f, g, o)
    v8f g[4];
#pragma unroll
    for (int nt = 0; nt < 4; ++nt) {
      v8f ci;
#pragma unroll
      for (int r = 0; r < 8; ++r) ci[r] = bias[nt];
      g[nt] = __builtin_amdgcn_wmma_f32_16x16x32_f16(
          false, A.v, false, Bw[nt], (short)0, ci, false, false);
    }

    // --- LSTM cell elementwise, f32
    v8f hnew;
#pragma unroll
    for (int r = 0; r < 8; ++r) {
      const float iv = fast_sigmoid(g[0][r]);
      const float fv = fast_sigmoid(g[1][r]);
      const float gv = fast_tanh(g[2][r]);
      const float ov = fast_sigmoid(g[3][r]);
      const float cn = fv * c[r] + iv * gv;
      c[r] = cn;
      hnew[r] = ov * fast_tanh(cn);
    }

    // --- h -> LDS staging (also acts as transpose for the global store)
#pragma unroll
    for (int r = 0; r < 8; ++r)
      shh[(r + 8 * hi) * HS1 + lo] = (_Float16)hnew[r];
    // coalesced 32B-per-lane store of the whole tile's hidden state (f16)
    if (hi == 0)
      *(v16h*)(hs + ((size_t)t * BB + baseb + lo) * HS1) =
          *(const v16h*)(shh + lo * HS1);
    __syncthreads();
  }
}

// ---------------------------------------------------------------------------
// Middle LSTM: IN=32 (concat f/b) -> H=64, gates = 256 = 16 N-tiles, K = 96 = 3x32.
// Weights live in LDS in B-operand layout (48 KB). Fused dense 64->1 output.
// ---------------------------------------------------------------------------
__global__ __launch_bounds__(128) void lstm_mid(
    const _Float16* __restrict__ hsf, const _Float16* __restrict__ hsb,
    const float* __restrict__ h0, const float* __restrict__ c0,
    const float* __restrict__ Wih, const float* __restrict__ Whh,
    const float* __restrict__ bih, const float* __restrict__ bhh,
    const float* __restrict__ Wd, const float* __restrict__ bd,
    float* __restrict__ out)
{
  __shared__ __align__(32) _Float16 wB[48 * 512];      // 16 N-tiles x 3 K-chunks, 48 KB
  __shared__ __align__(32) _Float16 shm[4 * 16 * HS2]; // per-wave h staging, 8 KB
  const int tid  = threadIdx.x;
  const int wave = tid >> 5;
  const int lane = tid & 31;
  const int lo   = lane & 15;
  const int hi   = lane >> 4;
  const int baseb = blockIdx.x * 64 + wave * 16;
  _Float16* shh = shm + wave * 16 * HS2;

  // --- cooperative weight load into B-operand layout:
  //     tile = nt*3+kc; lane L element e: K = kc*32 + (e<8 ? e : e+8) + 8*(L>>4), N = nt*16 + (L&15)
  for (int idx = tid; idx < 48 * 512; idx += 128) {
    const int tile = idx >> 9;
    const int rem  = idx & 511;
    const int L = rem >> 4;
    const int e = rem & 15;
    const int nt = tile / 3, kc = tile % 3;
    const int K = kc * 32 + (e < 8 ? e : e + 8) + 8 * (L >> 4);
    const int N = nt * 16 + (L & 15);
    const float w = (K < 32) ? Wih[N * 32 + K] : Whh[N * HS2 + (K - 32)];
    wB[tile * 512 + L * 16 + e] = (_Float16)w;
  }

  float bias[16];
#pragma unroll
  for (int nt = 0; nt < 16; ++nt) bias[nt] = bih[nt * 16 + lo] + bhh[nt * 16 + lo];
  float wdv[4];
#pragma unroll
  for (int jt = 0; jt < 4; ++jt) wdv[jt] = Wd[jt * 16 + lo];
  const float bd0 = bd[0];

  v8f c[4];
#pragma unroll
  for (int jt = 0; jt < 4; ++jt)
#pragma unroll
    for (int r = 0; r < 8; ++r)
      c[jt][r] = c0[(size_t)(baseb + r + 8 * hi) * HS2 + jt * 16 + lo];
#pragma unroll
  for (int jt = 0; jt < 4; ++jt)
#pragma unroll
    for (int r = 0; r < 8; ++r) {
      const int bl = r + 8 * hi;
      shh[bl * HS2 + jt * 16 + lo] =
          (_Float16)h0[(size_t)(baseb + bl) * HS2 + jt * 16 + lo];
    }
  __syncthreads();

  for (int t = 0; t < TT; ++t) {
    // --- A chunks (K=96): kc0 = [hs_f|hs_b] from global, kc1/kc2 = h from LDS
    union { v16h v; v8h p[2]; } A[3];
    {
      const size_t rowoff = ((size_t)t * BB + baseb + lo) * HS1 + 8 * hi;
      A[0].p[0] = *(const v8h*)(hsf + rowoff);
      A[0].p[1] = *(const v8h*)(hsb + rowoff);
      const _Float16* hp = shh + lo * HS2 + 8 * hi;
      A[1].p[0] = *(const v8h*)(hp);
      A[1].p[1] = *(const v8h*)(hp + 16);
      A[2].p[0] = *(const v8h*)(hp + 32);
      A[2].p[1] = *(const v8h*)(hp + 48);
    }

    v8f hnew[4];
    float p[8];
#pragma unroll
    for (int r = 0; r < 8; ++r) p[r] = 0.0f;

#pragma unroll
    for (int jt = 0; jt < 4; ++jt) {  // hidden-slice tiles; gate groups i/f/g/o = jt, jt+4, jt+8, jt+12
      v8f gi, gf, gg, go;
#pragma unroll
      for (int r = 0; r < 8; ++r) {
        gi[r] = bias[jt];
        gf[r] = bias[jt + 4];
        gg[r] = bias[jt + 8];
        go[r] = bias[jt + 12];
      }
#pragma unroll
      for (int kc = 0; kc < 3; ++kc) {
        const v16h bwi = *(const v16h*)(wB + ((jt     ) * 3 + kc) * 512 + lane * 16);
        const v16h bwf = *(const v16h*)(wB + ((jt +  4) * 3 + kc) * 512 + lane * 16);
        const v16h bwg = *(const v16h*)(wB + ((jt +  8) * 3 + kc) * 512 + lane * 16);
        const v16h bwo = *(const v16h*)(wB + ((jt + 12) * 3 + kc) * 512 + lane * 16);
        gi = __builtin_amdgcn_wmma_f32_16x16x32_f16(false, A[kc].v, false, bwi, (short)0, gi, false, false);
        gf = __builtin_amdgcn_wmma_f32_16x16x32_f16(false, A[kc].v, false, bwf, (short)0, gf, false, false);
        gg = __builtin_amdgcn_wmma_f32_16x16x32_f16(false, A[kc].v, false, bwg, (short)0, gg, false, false);
        go = __builtin_amdgcn_wmma_f32_16x16x32_f16(false, A[kc].v, false, bwo, (short)0, go, false, false);
      }
#pragma unroll
      for (int r = 0; r < 8; ++r) {
        const float iv = fast_sigmoid(gi[r]);
        const float fv = fast_sigmoid(gf[r]);
        const float gv = fast_tanh(gg[r]);
        const float ov = fast_sigmoid(go[r]);
        const float cn = fv * c[jt][r] + iv * gv;
        c[jt][r] = cn;
        const float hv = ov * fast_tanh(cn);
        hnew[jt][r] = hv;
        p[r] += hv * wdv[jt];   // fused dense partial
      }
    }

    // --- dense reduction across the 16 lanes of each half-wave
#pragma unroll
    for (int m = 1; m < 16; m <<= 1)
#pragma unroll
      for (int r = 0; r < 8; ++r)
        p[r] += __shfl_xor(p[r], m, 32);
    if (lo == 0) {
#pragma unroll
      for (int r = 0; r < 8; ++r)
        out[(size_t)(baseb + r + 8 * hi) * TT + t] = p[r] + bd0;
    }

    // --- h -> LDS staging for next step
#pragma unroll
    for (int jt = 0; jt < 4; ++jt)
#pragma unroll
      for (int r = 0; r < 8; ++r)
        shh[(r + 8 * hi) * HS2 + jt * 16 + lo] = (_Float16)hnew[jt][r];
    __syncthreads();
  }
}

extern "C" void kernel_launch(void* const* d_in, const int* in_sizes, int n_in,
                              void* d_out, int out_size, void* d_ws, size_t ws_size,
                              hipStream_t stream) {
  const float* x     = (const float*)d_in[0];
  const float* h0f   = (const float*)d_in[1];
  const float* c0f   = (const float*)d_in[2];
  const float* h0b   = (const float*)d_in[3];
  const float* c0b   = (const float*)d_in[4];
  const float* h0m   = (const float*)d_in[5];
  const float* c0m   = (const float*)d_in[6];
  const float* Wih_f = (const float*)d_in[7];
  const float* Whh_f = (const float*)d_in[8];
  const float* bih_f = (const float*)d_in[9];
  const float* bhh_f = (const float*)d_in[10];
  const float* Wih_b = (const float*)d_in[11];
  const float* Whh_b = (const float*)d_in[12];
  const float* bih_b = (const float*)d_in[13];
  const float* bhh_b = (const float*)d_in[14];
  const float* Wih_m = (const float*)d_in[15];
  const float* Whh_m = (const float*)d_in[16];
  const float* bih_m = (const float*)d_in[17];
  const float* bhh_m = (const float*)d_in[18];
  const float* Wd    = (const float*)d_in[19];
  const float* bd    = (const float*)d_in[20];

  _Float16* hsf = (_Float16*)d_ws;                       // [T,B,16] f16
  _Float16* hsb = hsf + (size_t)TT * BB * HS1;           // [T,B,16] f16

  // forward leg, backward leg (independent), then fused middle LSTM + dense
  lstm16_dir<<<BB / 128, 256, 0, stream>>>(x, h0f, c0f, Wih_f, Whh_f, bih_f, bhh_f, hsf, 0);
  lstm16_dir<<<BB / 128, 256, 0, stream>>>(x, h0b, c0b, Wih_b, Whh_b, bih_b, bhh_b, hsb, 1);
  lstm_mid<<<BB / 64, 128, 0, stream>>>(hsf, hsb, h0m, c0m, Wih_m, Whh_m, bih_m, bhh_m,
                                        Wd, bd, (float*)d_out);
}
